// RWKV_Tmix_x070_33492154974430
// MI455X (gfx1250) — matrix operand order repaired and checked by reading, not compiled, not run on hardware
//
#include <hip/hip_runtime.h>
#include <hip/hip_bf16.h>

// ---------------------------------------------------------------------------
// RWKV-7 Tmix for MI455X (gfx1250): bf16 WMMA GEMMs (async-LDS double
// buffered) + fused vector stages.
// ---------------------------------------------------------------------------

typedef __bf16 bf16;
typedef __attribute__((ext_vector_type(16))) __bf16 v16bf;
typedef __attribute__((ext_vector_type(8)))  float  v8f;
typedef int v4i __attribute__((vector_size(16)));   // matches builtin proto

constexpr int kB = 4;
constexpr int kT = 2048;
constexpr int kC = 2048;
constexpr int kH = 32;
constexpr int kN = 64;
constexpr int kM = kB * kT;             // 8192 rows
constexpr size_t kMC = (size_t)kM * kC; // 16,777,216 elements

// ---- CDNA5 async global->LDS path (guarded; falls back to load+ds_store) ---
#if defined(__has_builtin)
#  if __has_builtin(__builtin_amdgcn_global_load_async_to_lds_b128) && \
      __has_builtin(__builtin_amdgcn_s_wait_asynccnt)
#    define HAS_ASYNC_LDS 1
#  endif
#endif
#ifndef HAS_ASYNC_LDS
#  define HAS_ASYNC_LDS 0
#endif

#if HAS_ASYNC_LDS
typedef __attribute__((address_space(1))) v4i* g_v4i_ptr;
typedef __attribute__((address_space(3))) v4i* l_v4i_ptr;
#define ASYNC_COPY16(gsrc, ldst)                                        \
    __builtin_amdgcn_global_load_async_to_lds_b128(                     \
        (g_v4i_ptr)(gsrc), (l_v4i_ptr)(ldst), 0, 0)
#endif

// ===========================================================================
// bf16 WMMA GEMM: D[M,N] (f32) = A[M,K] (bf16 row-major, lda)
//                              x Bt[Npad,K] (bf16, PRE-TRANSPOSED weights)
// Block tile 128x128, 8 waves, each wave 32x64 = 2x4 WMMA(16x16x32) tiles.
// Double-buffered LDS; A and B tiles are identical 128x32 row-major copies.
// M multiple of 128; K multiple of 32; Npad = gridDim.x*128 (zero padded).
// ===========================================================================
#define GEMM_BM 128
#define GEMM_BN 128
#define GEMM_KC 32
#define GEMM_KP 48   // padded K pitch in LDS (16B-aligned rows, bank skew)

__global__ __launch_bounds__(256) void gemm_bf16_wmma(
    const bf16* __restrict__ A, int lda,
    const bf16* __restrict__ Bt, int ldb,    // ldb == K
    float* __restrict__ D, int ldd,
    int M, int N, int K)
{
    (void)M;
    __shared__ __align__(16) bf16 As[2][GEMM_BM][GEMM_KP];
    __shared__ __align__(16) bf16 Bs[2][GEMM_BN][GEMM_KP];  // Bs[buf][n][k]

    const int tid  = threadIdx.x;
    const int lane = tid & 31;
    const int wave = tid >> 5;
    const int wm   = wave >> 1;     // 0..3  (M direction)
    const int wn   = wave & 1;      // 0..1  (N direction)
    const int half = lane >> 4;     // lane half (wave32)
    const int lrow = lane & 15;

    const int bm0 = blockIdx.y * GEMM_BM;
    const int bn0 = blockIdx.x * GEMM_BN;

    // loader coordinates: 256 threads cover a 128x32 tile; 32B per thread
    const int trow = tid >> 1;           // 0..127
    const int tcg  = (tid & 1) * 16;     // 0 / 16

    const bf16* gA = A  + (size_t)(bm0 + trow) * lda + tcg;
    const bf16* gB = Bt + (size_t)(bn0 + trow) * ldb + tcg;

    v8f acc[2][4] = {};

    auto issue = [&](int k0, int buf) {
        const bf16* sa = gA + k0;
        const bf16* sb = gB + k0;
        bf16* da = &As[buf][trow][tcg];
        bf16* db = &Bs[buf][trow][tcg];
#if HAS_ASYNC_LDS
        ASYNC_COPY16(sa,     da);
        ASYNC_COPY16(sa + 8, da + 8);
        ASYNC_COPY16(sb,     db);
        ASYNC_COPY16(sb + 8, db + 8);
#else
        const uint4* s4a = reinterpret_cast<const uint4*>(sa);
        const uint4* s4b = reinterpret_cast<const uint4*>(sb);
        uint4 a0 = s4a[0], a1 = s4a[1];
        uint4 b0 = s4b[0], b1 = s4b[1];
        reinterpret_cast<uint4*>(da)[0] = a0;
        reinterpret_cast<uint4*>(da)[1] = a1;
        reinterpret_cast<uint4*>(db)[0] = b0;
        reinterpret_cast<uint4*>(db)[1] = b1;
#endif
    };

    auto compute = [&](int buf) {
        v16bf afr[2], bfr[4];
        #pragma unroll
        for (int tm = 0; tm < 2; ++tm) {
            const int r = wm * 32 + tm * 16 + lrow;
            union { uint4 u[2]; v16bf v; } t;
            // A frag: lane<16 -> K {0..7,16..23}; lane>=16 -> K {8..15,24..31}
            t.u[0] = *reinterpret_cast<const uint4*>(&As[buf][r][half * 8]);
            t.u[1] = *reinterpret_cast<const uint4*>(&As[buf][r][16 + half * 8]);
            afr[tm] = t.v;
        }
        #pragma unroll
        for (int tn = 0; tn < 4; ++tn) {
            const int cidx = wn * 64 + tn * 16 + lrow;   // column (n) index
            union { uint4 u[2]; v16bf v; } t;
            // B frag: lane<16 -> K {0..7,16..23}; lane>=16 -> K {8..15,24..31}
            t.u[0] = *reinterpret_cast<const uint4*>(&Bs[buf][cidx][half * 8]);
            t.u[1] = *reinterpret_cast<const uint4*>(&Bs[buf][cidx][16 + half * 8]);
            bfr[tn] = t.v;
        }
        #pragma unroll
        for (int tm = 0; tm < 2; ++tm)
            #pragma unroll
            for (int tn = 0; tn < 4; ++tn)
                acc[tm][tn] = __builtin_amdgcn_wmma_f32_16x16x32_bf16(
                    false, afr[tm], false, bfr[tn],
                    (short)0, acc[tm][tn], false, false);
    };

    const int nch = K / GEMM_KC;
    issue(0, 0);
    for (int ci = 0; ci < nch - 1; ++ci) {
        const int cur = ci & 1;
        issue((ci + 1) * GEMM_KC, cur ^ 1);
#if HAS_ASYNC_LDS
        __builtin_amdgcn_s_wait_asynccnt(4);  // previous chunk's 4 copies done
#endif
        __syncthreads();
        compute(cur);
        __syncthreads();
    }
    {
        const int cur = (nch - 1) & 1;
#if HAS_ASYNC_LDS
        __builtin_amdgcn_s_wait_asynccnt(0);
#endif
        __syncthreads();
        compute(cur);
    }

    // ---- epilogue: C/D layout — VGPR v, lane l: row = v + (l>>4)*8, col = l&15
    #pragma unroll
    for (int tm = 0; tm < 2; ++tm) {
        #pragma unroll
        for (int tn = 0; tn < 4; ++tn) {
            const int gcol = bn0 + wn * 64 + tn * 16 + lrow;
            if (gcol < N) {
                #pragma unroll
                for (int v = 0; v < 8; ++v) {
                    const int grow = bm0 + wm * 32 + tm * 16 + half * 8 + v;
                    D[(size_t)grow * ldd + gcol] = acc[tm][tn][v];
                }
            }
        }
    }
}

// ===========================================================================
// Stage 1: token shift + six lerps -> bf16 activation buffers
// ===========================================================================
__global__ __launch_bounds__(256) void prep_kernel(
    const float* __restrict__ x,
    const float* __restrict__ mr, const float* __restrict__ mw,
    const float* __restrict__ mk, const float* __restrict__ mv,
    const float* __restrict__ ma, const float* __restrict__ mg,
    bf16* __restrict__ xr, bf16* __restrict__ xw, bf16* __restrict__ xk,
    bf16* __restrict__ xv, bf16* __restrict__ xa, bf16* __restrict__ xg)
{
    const size_t e = (size_t)blockIdx.x * 256 + threadIdx.x;
    if (e >= kMC) return;
    const int c = (int)(e % kC);
    const size_t m = e / kC;
    const int t = (int)(m % kT);
    const float xc = x[e];
    const float xp = (t == 0) ? 0.0f : x[e - kC];
    const float xx = xp - xc;
    xr[e] = (bf16)(xc + xx * mr[c]);
    xw[e] = (bf16)(xc + xx * mw[c]);
    xk[e] = (bf16)(xc + xx * mk[c]);
    xv[e] = (bf16)(xc + xx * mv[c]);
    xa[e] = (bf16)(xc + xx * ma[c]);
    xg[e] = (bf16)(xc + xx * mg[c]);
}

// ===========================================================================
// f32 -> bf16 convert, optional activation (0=copy, 1=tanh, 2=sigmoid)
// ===========================================================================
__global__ __launch_bounds__(256) void cvt_kernel(
    const float* __restrict__ src, bf16* __restrict__ dst, size_t n, int mode)
{
    const size_t i = (size_t)blockIdx.x * 256 + threadIdx.x;
    if (i >= n) return;
    float v = src[i];
    if (mode == 1)      v = tanhf(v);
    else if (mode == 2) v = 1.0f / (1.0f + expf(-v));
    dst[i] = (bf16)v;
}

// ===========================================================================
// Weight convert + transpose: src f32 [K][N] -> dst bf16 [Npad][K] with
// zero padding for n >= N (so the GEMM never needs N-masked tile loads).
// ===========================================================================
__global__ __launch_bounds__(256) void cvt_transpose_kernel(
    const float* __restrict__ src, bf16* __restrict__ dst,
    int K, int N, int Npad)
{
    const size_t i = (size_t)blockIdx.x * 256 + threadIdx.x;
    if (i >= (size_t)Npad * K) return;
    const int n = (int)(i / K);
    const int k = (int)(i % K);
    dst[i] = (n < N) ? (bf16)src[(size_t)k * N + n] : (bf16)0.0f;
}

// ===========================================================================
// Stage 4: fused decay / value-residual / in-context-lr / key-norm.
// 256 threads = 4 head-rows of 64; per-head L2 norm via LDS reduction.
// In-place: W<-exp(-exp(w)), V<-v_final, K<-k_final, A<- -kk_n, Bf<- kk_n*a
// ===========================================================================
__global__ __launch_bounds__(256) void fuse_pre_kernel(
    float* __restrict__ Kf, float* __restrict__ Wf, float* __restrict__ Vf,
    float* __restrict__ Af, float* __restrict__ Bf,
    const float* __restrict__ v0,
    const float* __restrict__ time_decay, const float* __restrict__ time_aaaaa,
    const float* __restrict__ time_misc_v, const float* __restrict__ time_misc_kkk,
    const float* __restrict__ time_misc_a)
{
    __shared__ float red[256];
    const int tid = threadIdx.x;
    const int seg = tid >> 6;
    const int n   = tid & 63;
    const size_t row = (size_t)blockIdx.x * 4 + seg;     // in [0, M*H)
    const size_t m = row / kH;
    const int h = (int)(row % kH);
    const int c = h * kN + n;
    const size_t off = m * kC + c;

    const float kv  = Kf[off];
    const float wmx = Wf[off];
    const float vr  = Vf[off];
    const float ap  = Af[off];
    const float mvp = Bf[off];

    const float z  = time_decay[c] + wmx;
    const float w  = -log1pf(expf(-z)) - 0.5f;           // -softplus(-z) - 0.5
    const float wd = expf(-expf(w));
    const float sv = 1.0f / (1.0f + expf(-(time_misc_v[c] + mvp)));
    const float vf = vr + (v0[off] - vr) * sv;
    const float a  = 1.0f / (1.0f + expf(-(time_aaaaa[c] + ap)));
    const float kk = kv * time_misc_kkk[c];

    red[tid] = kk * kk;
    __syncthreads();
    #pragma unroll
    for (int s = 32; s > 0; s >>= 1) {
        if (n < s) red[tid] += red[tid + s];
        __syncthreads();
    }
    const float inv = 1.0f / fmaxf(sqrtf(red[seg * 64]), 1e-12f);
    const float kkn = kk * inv;

    Wf[off] = wd;
    Vf[off] = vf;
    Af[off] = -kkn;
    Bf[off] = kkn * a;
    Kf[off] = kv * (1.0f + (a - 1.0f) * time_misc_a[c]);
}

// ===========================================================================
// Stage 5: WKV7 recurrence. One 64-thread block per (b,h); thread i owns
// state row i (64 VGPRs). Per step: broadcast r,w,k,v,a,b via LDS, then
// per-thread dot + fused update + output dot. Sequential over T=2048.
// ===========================================================================
__global__ __launch_bounds__(64) void wkv7_kernel(
    const float* __restrict__ R,  const float* __restrict__ Wd,
    const float* __restrict__ Kf, const float* __restrict__ Vf,
    const float* __restrict__ Aa, const float* __restrict__ Bb,
    float* __restrict__ Y)
{
    const int bh = blockIdx.x;              // b*H + h
    const int b = bh / kH, h = bh % kH;
    const int i = threadIdx.x;
    __shared__ float sr[64], sw[64], sk[64], sv[64], sa[64], sb[64];

    float st[64];
    #pragma unroll
    for (int j = 0; j < 64; ++j) st[j] = 0.0f;

    const size_t base = (size_t)b * kT * kC + (size_t)h * kN + i;
    for (int t = 0; t < kT; ++t) {
        const size_t off = base + (size_t)t * kC;
        sr[i] = R[off];  sw[i] = Wd[off];
        sk[i] = Kf[off]; sv[i] = Vf[off];
        sa[i] = Aa[off]; sb[i] = Bb[off];
        __syncthreads();

        float sad = 0.0f;
        #pragma unroll
        for (int j = 0; j < 64; ++j) sad += st[j] * sa[j];
        const float vi = sv[i];
        float y = 0.0f;
        #pragma unroll
        for (int j = 0; j < 64; ++j) {
            const float s = st[j] * sw[j] + sad * sb[j] + vi * sk[j];
            st[j] = s;
            y += s * sr[j];
        }
        Y[off] = y;
        __syncthreads();
    }
}

// ===========================================================================
// Stage 6: per-head GroupNorm + bonus + gating -> bf16 for final GEMM.
// ===========================================================================
__global__ __launch_bounds__(256) void gn_bonus_gate_kernel(
    const float* __restrict__ Y, const float* __restrict__ R,
    const float* __restrict__ Kf, const float* __restrict__ Vf,
    const float* __restrict__ G,
    const float* __restrict__ ln_w, const float* __restrict__ ln_b,
    const float* __restrict__ faaaa,
    bf16* __restrict__ Z)
{
    __shared__ float red[256];
    const int tid = threadIdx.x;
    const int seg = tid >> 6;
    const int n   = tid & 63;
    const size_t row = (size_t)blockIdx.x * 4 + seg;
    const size_t m = row / kH;
    const int h = (int)(row % kH);
    const int c = h * kN + n;
    const size_t off = m * kC + c;

    const float y = Y[off];

    red[tid] = y;
    __syncthreads();
    #pragma unroll
    for (int s = 32; s > 0; s >>= 1) { if (n < s) red[tid] += red[tid + s]; __syncthreads(); }
    const float mu = red[seg * 64] * (1.0f / 64.0f);
    __syncthreads();

    const float dy = y - mu;
    red[tid] = dy * dy;
    __syncthreads();
    #pragma unroll
    for (int s = 32; s > 0; s >>= 1) { if (n < s) red[tid] += red[tid + s]; __syncthreads(); }
    const float var = red[seg * 64] * (1.0f / 64.0f);
    __syncthreads();

    const float yn = dy * rsqrtf(var + 6.4e-4f) * ln_w[c] + ln_b[c];

    red[tid] = R[off] * Kf[off] * faaaa[c];
    __syncthreads();
    #pragma unroll
    for (int s = 32; s > 0; s >>= 1) { if (n < s) red[tid] += red[tid + s]; __syncthreads(); }
    const float dot = red[seg * 64];

    Z[off] = (bf16)((yn + dot * Vf[off]) * G[off]);
}

// ===========================================================================
// Host-side launch
// ===========================================================================
extern "C" void kernel_launch(void* const* d_in, const int* in_sizes, int n_in,
                              void* d_out, int out_size, void* d_ws, size_t ws_size,
                              hipStream_t stream)
{
    (void)in_sizes; (void)n_in; (void)out_size; (void)ws_size;

    const float* x        = (const float*)d_in[0];
    const float* v0       = (const float*)d_in[1];
    const float* t_maa_r  = (const float*)d_in[2];
    const float* t_maa_w  = (const float*)d_in[3];
    const float* t_maa_k  = (const float*)d_in[4];
    const float* t_maa_v  = (const float*)d_in[5];
    const float* t_maa_a  = (const float*)d_in[6];
    const float* t_maa_g  = (const float*)d_in[7];
    const float* t_decay  = (const float*)d_in[8];
    const float* t_faaaa  = (const float*)d_in[9];
    const float* t_aaaaa  = (const float*)d_in[10];
    const float* td_w1    = (const float*)d_in[11];
    const float* td_w2    = (const float*)d_in[12];
    const float* aaa_w1   = (const float*)d_in[13];
    const float* aaa_w2   = (const float*)d_in[14];
    const float* gate_w1  = (const float*)d_in[15];
    const float* gate_w2  = (const float*)d_in[16];
    const float* mv_w1    = (const float*)d_in[17];
    const float* mv_w2    = (const float*)d_in[18];
    const float* t_misc_v = (const float*)d_in[19];
    const float* t_misc_k = (const float*)d_in[20];
    const float* t_misc_a = (const float*)d_in[21];
    const float* Wr       = (const float*)d_in[22];
    const float* Wk       = (const float*)d_in[23];
    const float* Wv       = (const float*)d_in[24];
    const float* Wo       = (const float*)d_in[25];
    const float* ln_w     = (const float*)d_in[26];
    const float* ln_b     = (const float*)d_in[27];

    float* out = (float*)d_out;

    // ---- workspace carve-out (256B aligned)
    char* ws = (char*)d_ws;
    size_t off = 0;
    auto alloc = [&](size_t bytes) -> void* {
        void* p = ws + off;
        off += (bytes + 255) & ~(size_t)255;
        return p;
    };
    const size_t CC = (size_t)kC * kC;

    // bf16 activation buffers
    bf16* xr = (bf16*)alloc(kMC * 2);
    bf16* xw = (bf16*)alloc(kMC * 2);
    bf16* xk = (bf16*)alloc(kMC * 2);
    bf16* xv = (bf16*)alloc(kMC * 2);
    bf16* xa = (bf16*)alloc(kMC * 2);
    bf16* xg = (bf16*)alloc(kMC * 2);
    bf16* zb = (bf16*)alloc(kMC * 2);
    // bf16 transposed weights [Npad][K]
    bf16* wrT   = (bf16*)alloc(CC * 2);                    // [2048][2048]
    bf16* wkT   = (bf16*)alloc(CC * 2);
    bf16* wvT   = (bf16*)alloc(CC * 2);
    bf16* woT   = (bf16*)alloc(CC * 2);
    bf16* tdw1T = (bf16*)alloc((size_t)128 * kC * 2);      // [128][2048] (N=64)
    bf16* tdw2T = (bf16*)alloc((size_t)kC * 64 * 2);       // [2048][64]
    bf16* aaw1T = (bf16*)alloc((size_t)128 * kC * 2);
    bf16* aaw2T = (bf16*)alloc((size_t)kC * 64 * 2);
    bf16* gw1T  = (bf16*)alloc((size_t)128 * kC * 2);      // [128][2048] (N=128)
    bf16* gw2T  = (bf16*)alloc((size_t)kC * 128 * 2);      // [2048][128]
    bf16* mvw1T = (bf16*)alloc((size_t)128 * kC * 2);      // [128][2048] (N=32)
    bf16* mvw2T = (bf16*)alloc((size_t)kC * 32 * 2);       // [2048][32]
    // f32 intermediates
    float* Rf  = (float*)alloc(kMC * 4);
    float* Kf  = (float*)alloc(kMC * 4);
    float* Vf  = (float*)alloc(kMC * 4);
    float* Wf  = (float*)alloc(kMC * 4);
    float* Af  = (float*)alloc(kMC * 4);
    float* MVf = (float*)alloc(kMC * 4);
    float* Gf  = (float*)alloc(kMC * 4);
    float* Yf  = (float*)alloc(kMC * 4);
    // small MLP hidden buffers
    float* htd_f = (float*)alloc((size_t)kM * 64 * 4);
    float* haa_f = (float*)alloc((size_t)kM * 64 * 4);
    float* hmv_f = (float*)alloc((size_t)kM * 32 * 4);
    float* hg_f  = (float*)alloc((size_t)kM * 128 * 4);
    bf16* htd_b = (bf16*)alloc((size_t)kM * 64 * 2);
    bf16* haa_b = (bf16*)alloc((size_t)kM * 64 * 2);
    bf16* hmv_b = (bf16*)alloc((size_t)kM * 32 * 2);
    bf16* hg_b  = (bf16*)alloc((size_t)kM * 128 * 2);

    auto cvt = [&](const float* s, bf16* d, size_t n, int mode) {
        cvt_kernel<<<(unsigned)((n + 255) / 256), 256, 0, stream>>>(s, d, n, mode);
    };
    auto cvtT = [&](const float* s, bf16* d, int K, int N, int Npad) {
        size_t n = (size_t)Npad * K;
        cvt_transpose_kernel<<<(unsigned)((n + 255) / 256), 256, 0, stream>>>(
            s, d, K, N, Npad);
    };
    auto gemm = [&](const bf16* A, int lda, const bf16* Bm, int ldb,
                    float* D, int ldd, int M, int N, int K) {
        dim3 grid((N + GEMM_BN - 1) / GEMM_BN, M / GEMM_BM);
        gemm_bf16_wmma<<<grid, 256, 0, stream>>>(A, lda, Bm, ldb, D, ldd, M, N, K);
    };

    // Stage 0: weight conversion to bf16, transposed + padded
    cvtT(Wr, wrT, kC, kC, kC);
    cvtT(Wk, wkT, kC, kC, kC);
    cvtT(Wv, wvT, kC, kC, kC);
    cvtT(Wo, woT, kC, kC, kC);
    cvtT(td_w1,  tdw1T, kC, 64, 128);
    cvtT(td_w2,  tdw2T, 64, kC, kC);
    cvtT(aaa_w1, aaw1T, kC, 64, 128);
    cvtT(aaa_w2, aaw2T, 64, kC, kC);
    cvtT(gate_w1, gw1T, kC, 128, 128);
    cvtT(gate_w2, gw2T, 128, kC, kC);
    cvtT(mv_w1,  mvw1T, kC, 32, 128);
    cvtT(mv_w2,  mvw2T, 32, kC, kC);

    // Stage 1: token shift + lerps
    prep_kernel<<<(unsigned)(kMC / 256), 256, 0, stream>>>(
        x, t_maa_r, t_maa_w, t_maa_k, t_maa_v, t_maa_a, t_maa_g,
        xr, xw, xk, xv, xa, xg);

    // Stage 2: big projections (WMMA)
    gemm(xr, kC, wrT, kC, Rf, kC, kM, kC, kC);
    gemm(xk, kC, wkT, kC, Kf, kC, kM, kC, kC);
    gemm(xv, kC, wvT, kC, Vf, kC, kM, kC, kC);

    // Stage 3: small MLPs (WMMA; w1 outputs padded to 128 cols, store-masked)
    gemm(xw, kC, tdw1T, kC, htd_f, 64, kM, 64, kC);
    cvt(htd_f, htd_b, (size_t)kM * 64, /*tanh*/1);
    gemm(htd_b, 64, tdw2T, 64, Wf, kC, kM, kC, 64);

    gemm(xa, kC, aaw1T, kC, haa_f, 64, kM, 64, kC);
    cvt(haa_f, haa_b, (size_t)kM * 64, 0);
    gemm(haa_b, 64, aaw2T, 64, Af, kC, kM, kC, 64);

    gemm(xv, kC, mvw1T, kC, hmv_f, 32, kM, 32, kC);
    cvt(hmv_f, hmv_b, (size_t)kM * 32, 0);
    gemm(hmv_b, 32, mvw2T, 32, MVf, kC, kM, kC, 32);

    gemm(xg, kC, gw1T, kC, hg_f, 128, kM, 128, kC);
    cvt(hg_f, hg_b, (size_t)kM * 128, /*sigmoid*/2);
    gemm(hg_b, 128, gw2T, 128, Gf, kC, kM, kC, 128);

    // Stage 4: fused pre-recurrence elementwise + per-head key normalization
    fuse_pre_kernel<<<(unsigned)((size_t)kM * kH / 4), 256, 0, stream>>>(
        Kf, Wf, Vf, Af, MVf, v0, t_decay, t_aaaaa, t_misc_v, t_misc_k, t_misc_a);

    // Stage 5: WKV7 recurrence
    wkv7_kernel<<<kB * kH, 64, 0, stream>>>(Rf, Wf, Kf, Vf, Af, MVf, Yf);

    // Stage 6: GroupNorm + bonus + gate -> bf16
    gn_bonus_gate_kernel<<<(unsigned)((size_t)kM * kH / 4), 256, 0, stream>>>(
        Yf, Rf, Kf, Vf, Gf, ln_w, ln_b, t_faaaa, zb);

    // Stage 7: output projection (WMMA) -> first half of d_out
    gemm(zb, kC, woT, kC, out, kC, kM, kC, kC);

    // Stage 8: second tuple element: v0 passthrough
    (void)hipMemcpyAsync(out + kMC, v0, kMC * sizeof(float),
                         hipMemcpyDeviceToDevice, stream);
}
